// TransformerNodeEncoder_79156247265635
// MI455X (gfx1250) — matrix operand order repaired and checked by reading, not compiled, not run on hardware
//
#include <hip/hip_runtime.h>

// Problem constants (deterministic from the reference): S=256 max nodes, B=128,
// D=256, H=8, dh=32, F=1024, L=2, tokens = S*B = 32768, token index = s*B + b.
#define SEQ   256
#define NB    128
#define DM    256
#define HEADS 8
#define DH    32
#define FF    1024
#define NTOK  32768

typedef _Float16 v8h  __attribute__((ext_vector_type(8)));
typedef _Float16 v16h __attribute__((ext_vector_type(16)));
typedef float    v8f  __attribute__((ext_vector_type(8)));
typedef float    v4f  __attribute__((ext_vector_type(4)));
typedef unsigned int v4u __attribute__((ext_vector_type(4)));
typedef int      v4i  __attribute__((ext_vector_type(4)));
typedef int      v8i  __attribute__((ext_vector_type(8)));

#if defined(__has_builtin)
#if __has_builtin(__builtin_amdgcn_tensor_load_to_lds) && \
    __has_builtin(__builtin_amdgcn_s_wait_tensorcnt)
#define USE_TDM 1
#endif
#endif
#ifndef USE_TDM
#define USE_TDM 0
#endif

static __device__ __forceinline__ v16h cat8(v8h lo, v8h hi) {
  v16h r;
#pragma unroll
  for (int j = 0; j < 8; ++j) { r[j] = lo[j]; r[8 + j] = hi[j]; }
  return r;
}
// A-operand fragment (16x32 f16). Lane holds row m; lane-half hl selects K sets:
// VGPR0-3: K = 8*hl + 0..7 ; VGPR4-7: K = 16 + 8*hl + 0..7   (ISA 7.12.2)
static __device__ __forceinline__ v16h fragA(const _Float16* row, int hl) {
  v8h lo = *(const v8h*)(row + 8 * hl);
  v8h hi = *(const v8h*)(row + 16 + 8 * hl);
  return cat8(lo, hi);
}
// B-operand fragment (32x16 f16). Lane holds column n (= a K-major source row);
// K = 8*hl + 0..7, then 16 + 8*hl + 0..7 (same as A).
static __device__ __forceinline__ v16h fragB(const _Float16* row, int hl) {
  v8h lo = *(const v8h*)(row + 8 * hl);
  v8h hi = *(const v8h*)(row + 16 + 8 * hl);
  return cat8(lo, hi);
}
static __device__ __forceinline__ v8f wmma16(v16h a, v16h b, v8f c) {
  return __builtin_amdgcn_wmma_f32_16x16x32_f16(false, a, false, b, (short)0, c,
                                                false, false);
}

#if USE_TDM
// TDM: load a 2-D f16 tile (tile_rows x tile_cols elements, row stride
// row_stride_elems) from global to LDS, padding the LDS destination by
// 16 bytes after every 64 bytes stored => LDS row stride = tile_cols(32)+8
// halfs = the 40-half padded layout used by the fragment loaders.
// D# layout per CDNA5 ISA ch.8 (groups 2/3 zero => plain 2-D tile).
static __device__ __forceinline__ void tdm_load_2d_f16(
    const _Float16* gsrc, unsigned lds_off, unsigned tile_cols,
    unsigned tile_rows, unsigned long long row_stride_elems) {
  unsigned long long ga = (unsigned long long)(const void*)gsrc;
  v4u g0;
  g0[0] = 1u;                                   // count=1 (valid), user mode
  g0[1] = lds_off;                              // lds_addr [63:32]
  g0[2] = (unsigned)(ga & 0xffffffffull);       // global_addr[31:0]
  g0[3] = (unsigned)((ga >> 32) & 0x1ffffffull) | (2u << 30);  // addr hi,type=2
  v8i g1;
  // [17:16] data_size=1 (2B); [20] pad_enable; [24:22] pad_interval=3 (16 DW);
  // [31:25] pad_amount=3 (4 DW)
  g1[0] = (1 << 16) | (1 << 20) | (3 << 22) | (3 << 25);
  unsigned td0 = (unsigned)row_stride_elems;    // tensor_dim0 (>= tile_cols)
  unsigned td1 = 1u << 22;                      // tensor_dim1 (large)
  g1[1] = (int)((td0 & 0xffffu) << 16);         // [63:48]=dim0 lo
  g1[2] = (int)((td0 >> 16) | ((td1 & 0xffffu) << 16));  // dim0 hi | dim1 lo
  g1[3] = (int)((td1 >> 16) | (tile_cols << 16));        // dim1 hi | tile_dim0
  g1[4] = (int)(tile_rows & 0xffffu);           // tile_dim1 (tile_dim2 = 0)
  g1[5] = (int)(row_stride_elems & 0xffffffffull);       // dim0_stride lo
  g1[6] = (int)((row_stride_elems >> 32) & 0xffffull);   // dim0_stride hi
  g1[7] = 0;
  v4i zz4 = {0, 0, 0, 0};
  v8i zz8 = {0, 0, 0, 0, 0, 0, 0, 0};
  // 6-arg toolchain form: (g0, g1, g2, g3, g4, cpol)
  __builtin_amdgcn_tensor_load_to_lds(g0, g1, zz4, zz4, zz8, 0);
}
#endif

// ---------------------------------------------------------------- utilities
__global__ void zero_f32(float* p, size_t n) {
  size_t i = (size_t)blockIdx.x * blockDim.x + threadIdx.x;
  size_t st = (size_t)gridDim.x * blockDim.x;
  for (; i < n; i += st) p[i] = 0.f;
}
__global__ void zero_i32(int* p, int n) {
  int i = blockIdx.x * blockDim.x + threadIdx.x;
  if (i < n) p[i] = 0;
}
__global__ void cvt_f16(const float* __restrict__ s, _Float16* __restrict__ d,
                        size_t n) {
  size_t i = (size_t)blockIdx.x * blockDim.x + threadIdx.x;
  size_t st = (size_t)gridDim.x * blockDim.x;
  for (; i < n; i += st) d[i] = (_Float16)s[i];
}
__global__ void count_kernel(const int* __restrict__ batch, int* counts,
                             int total) {
  int i = blockIdx.x * blockDim.x + threadIdx.x;
  if (i < total) atomicAdd(&counts[batch[i]], 1);
}
__global__ void scan_kernel(const int* __restrict__ counts, int* starts) {
  if (threadIdx.x == 0) {
    int acc = 0;
    for (int b = 0; b < NB; ++b) { starts[b] = acc; acc += counts[b]; }
  }
}
// front padding: row = (S - count[b]) + (node - start[b]); token = row*B + b
__global__ void scatter_kernel(const float* __restrict__ h,
                               const int* __restrict__ batch,
                               const int* __restrict__ counts,
                               const int* __restrict__ starts,
                               float* __restrict__ X, _Float16* __restrict__ Xh,
                               int total) {
  int idx = blockIdx.x * blockDim.x + threadIdx.x;
  if (idx >= total * DM) return;
  int node = idx >> 8, d = idx & 255;
  int b = batch[node];
  int row = (SEQ - counts[b]) + (node - starts[b]);
  float v = h[(size_t)node * DM + d];
  size_t o = ((size_t)(row * NB + b)) * DM + d;
  X[o] = v;
  Xh[o] = (_Float16)v;
}

// ------------------------------------------------- tiled WMMA GEMM (C=A*W^T+b)
// A: [M,K] f16, W: [N,K] f16 row-major (K-major => fragment friendly).
// Block tile 128x128, 8 waves in 4x2, wave tile 32x64, BK=32.
// Staging via Tensor Data Mover (TENSOR_LOAD_TO_LDS + s_wait_tensorcnt).
template <bool OUTF16, bool RELU>
__global__ void __launch_bounds__(256) gemm_wmma(
    const _Float16* __restrict__ A, const _Float16* __restrict__ W,
    const float* __restrict__ bias, float* __restrict__ outF,
    _Float16* __restrict__ outH, int M, int N, int K) {
  __shared__ _Float16 As[128][40];
  __shared__ _Float16 Bs[128][40];
  const int m0 = blockIdx.x * 128, n0 = blockIdx.y * 128;
  const int t = threadIdx.x, wid = t >> 5, lane = t & 31;
  const int ln = lane & 15, hl = lane >> 4;
  const int mW = (wid >> 1) * 32, nW = (wid & 1) * 64;

  v8f acc[2][4];
#pragma unroll
  for (int i = 0; i < 2; ++i)
#pragma unroll
    for (int j = 0; j < 4; ++j)
      acc[i][j] = (v8f){0.f, 0.f, 0.f, 0.f, 0.f, 0.f, 0.f, 0.f};

#if USE_TDM
  const unsigned ldsA = (unsigned)(size_t)&As[0][0];
  const unsigned ldsB = (unsigned)(size_t)&Bs[0][0];
#else
  const int sr = t >> 1, sc = (t & 1) * 16;
#endif

  for (int k0 = 0; k0 < K; k0 += 32) {
#if USE_TDM
    if (wid == 0) {
      tdm_load_2d_f16(A + (size_t)m0 * K + k0, ldsA, 32, 128,
                      (unsigned long long)K);
      tdm_load_2d_f16(W + (size_t)n0 * K + k0, ldsB, 32, 128,
                      (unsigned long long)K);
      __builtin_amdgcn_s_wait_tensorcnt(0);
    }
#else
    {
      const _Float16* srcA = A + (size_t)(m0 + sr) * K + k0 + sc;
      *(v8h*)&As[sr][sc] = *(const v8h*)srcA;
      *(v8h*)&As[sr][sc + 8] = *(const v8h*)(srcA + 8);
      const _Float16* srcB = W + (size_t)(n0 + sr) * K + k0 + sc;
      *(v8h*)&Bs[sr][sc] = *(const v8h*)srcB;
      *(v8h*)&Bs[sr][sc + 8] = *(const v8h*)(srcB + 8);
      if (k0 + 32 < K) {
        __builtin_prefetch(srcA + 32, 0, 1);
        __builtin_prefetch(srcB + 32, 0, 1);
      }
    }
#endif
    __syncthreads();
    v16h a[2], b[4];
#pragma unroll
    for (int mf = 0; mf < 2; ++mf) a[mf] = fragA(&As[mW + mf * 16 + ln][0], hl);
#pragma unroll
    for (int nf = 0; nf < 4; ++nf) b[nf] = fragB(&Bs[nW + nf * 16 + ln][0], hl);
#pragma unroll
    for (int mf = 0; mf < 2; ++mf)
#pragma unroll
      for (int nf = 0; nf < 4; ++nf)
        acc[mf][nf] = wmma16(a[mf], b[nf], acc[mf][nf]);
    __syncthreads();
  }
#pragma unroll
  for (int mf = 0; mf < 2; ++mf)
#pragma unroll
    for (int nf = 0; nf < 4; ++nf) {
      int nG = n0 + nW + nf * 16 + ln;
      float bv = bias[nG];
#pragma unroll
      for (int r = 0; r < 8; ++r) {
        int mG = m0 + mW + mf * 16 + r + 8 * hl;  // C layout: M = r + 8*hl
        float v = acc[mf][nf][r] + bv;
        if (RELU) v = fmaxf(v, 0.f);
        if (OUTF16) outH[(size_t)mG * N + nG] = (_Float16)v;
        else        outF[(size_t)mG * N + nG] = v;
      }
    }
}

// ------------------------------------------------------- attention per (b,h)
// QKV: f16 [NTOK, 3*DM]; O: f16 [NTOK, DM]. One 4-wave block per (b,h).
__global__ void __launch_bounds__(128) attn_kernel(
    const _Float16* __restrict__ QKV, const int* __restrict__ counts,
    _Float16* __restrict__ O) {
  __shared__ _Float16 Qs[SEQ][40];
  __shared__ _Float16 Ks[SEQ][40];
  __shared__ _Float16 Vt[DH][264];       // V transposed: Vt[d][t]
  __shared__ _Float16 Pw[4][16][272];    // per-wave probs tile (unnormalized)
  const int b = blockIdx.x >> 3, h = blockIdx.x & 7;
  const int t = threadIdx.x, wid = t >> 5, lane = t & 31;
  const int ln = lane & 15, hl = lane >> 4;

#if USE_TDM
  // TDM gathers the strided Q/K rows (token s*NB+b => row stride NB*3*DM elems)
  if (wid == 0) {
    const size_t base = (size_t)b * (3 * DM) + (size_t)h * DH;
    tdm_load_2d_f16(QKV + base, (unsigned)(size_t)&Qs[0][0], 32, SEQ,
                    (unsigned long long)NB * (3 * DM));
    tdm_load_2d_f16(QKV + base + DM, (unsigned)(size_t)&Ks[0][0], 32, SEQ,
                    (unsigned long long)NB * (3 * DM));
    __builtin_amdgcn_s_wait_tensorcnt(0);
  }
#pragma unroll
  for (int p = 0; p < 4; ++p) {  // V needs a transpose: stage manually
    int s = p * 64 + (t >> 1);
    int cs = (t & 1) * 16;
    const _Float16* base = QKV + (size_t)(s * NB + b) * (3 * DM) + h * DH + cs;
    v8h v0 = *(const v8h*)(base + 2 * DM);
    v8h v1 = *(const v8h*)(base + 2 * DM + 8);
#pragma unroll
    for (int j = 0; j < 8; ++j) {
      Vt[cs + j][s] = v0[j];
      Vt[cs + 8 + j][s] = v1[j];
    }
  }
#else
#pragma unroll
  for (int p = 0; p < 4; ++p) {
    int s = p * 64 + (t >> 1);
    int cs = (t & 1) * 16;
    const _Float16* base = QKV + (size_t)(s * NB + b) * (3 * DM) + h * DH + cs;
    *(v8h*)&Qs[s][cs] = *(const v8h*)base;
    *(v8h*)&Qs[s][cs + 8] = *(const v8h*)(base + 8);
    *(v8h*)&Ks[s][cs] = *(const v8h*)(base + DM);
    *(v8h*)&Ks[s][cs + 8] = *(const v8h*)(base + DM + 8);
    v8h v0 = *(const v8h*)(base + 2 * DM);
    v8h v1 = *(const v8h*)(base + 2 * DM + 8);
#pragma unroll
    for (int j = 0; j < 8; ++j) {
      Vt[cs + j][s] = v0[j];
      Vt[cs + 8 + j][s] = v1[j];
    }
  }
#endif
  __syncthreads();
  const int pad = SEQ - counts[b];
  const float scale = 0.17677669529663687f;  // 1/sqrt(32)

  for (int qt = wid; qt < 16; qt += 4) {
    v16h aq = fragA(&Qs[qt * 16 + ln][0], hl);
    float rm[8], rs[8];
#pragma unroll
    for (int r = 0; r < 8; ++r) { rm[r] = -3.0e38f; rs[r] = 0.f; }
    const v8f z = (v8f){0.f, 0.f, 0.f, 0.f, 0.f, 0.f, 0.f, 0.f};
    // pass 1: row max (scores = Q.K^T * scale + key mask bias)
    for (int kt = 0; kt < 16; ++kt) {
      v16h bk = fragB(&Ks[kt * 16 + ln][0], hl);
      v8f c = wmma16(aq, bk, z);
      float bias = (kt * 16 + ln < pad) ? -1.0e9f : 0.f;
#pragma unroll
      for (int r = 0; r < 8; ++r) rm[r] = fmaxf(rm[r], c[r] * scale + bias);
    }
#pragma unroll
    for (int msk = 1; msk < 16; msk <<= 1)
#pragma unroll
      for (int r = 0; r < 8; ++r)
        rm[r] = fmaxf(rm[r], __shfl_xor(rm[r], msk, 16));
    // pass 2: exp, row-sum, store unnormalized probs (<=1, f16-safe)
    for (int kt = 0; kt < 16; ++kt) {
      v16h bk = fragB(&Ks[kt * 16 + ln][0], hl);
      v8f c = wmma16(aq, bk, z);
      float bias = (kt * 16 + ln < pad) ? -1.0e9f : 0.f;
#pragma unroll
      for (int r = 0; r < 8; ++r) {
        float p = __expf(c[r] * scale + bias - rm[r]);
        rs[r] += p;
        Pw[wid][r + 8 * hl][kt * 16 + ln] = (_Float16)p;
      }
    }
#pragma unroll
    for (int msk = 1; msk < 16; msk <<= 1)
#pragma unroll
      for (int r = 0; r < 8; ++r) rs[r] += __shfl_xor(rs[r], msk, 16);
    float inv[8];
#pragma unroll
    for (int r = 0; r < 8; ++r) inv[r] = 1.f / rs[r];
    // O = P @ V, 8 k-steps of 32 over t
    v8f o0 = z, o1 = z;
#pragma unroll
    for (int ks = 0; ks < 8; ++ks) {
      v16h ap = fragA(&Pw[wid][ln][ks * 32], hl);
      v16h b0 = fragB(&Vt[ln][ks * 32], hl);        // d = ln
      v16h b1 = fragB(&Vt[16 + ln][ks * 32], hl);   // d = 16 + ln
      o0 = wmma16(ap, b0, o0);
      o1 = wmma16(ap, b1, o1);
    }
#pragma unroll
    for (int r = 0; r < 8; ++r) {
      int s = qt * 16 + r + 8 * hl;
      size_t base = (size_t)(s * NB + b) * DM + h * DH;
      O[base + ln] = (_Float16)(o0[r] * inv[r]);
      O[base + 16 + ln] = (_Float16)(o1[r] * inv[r]);
    }
  }
}

// ---------------------------------------------- LayerNorm (wave per row, D=256)
__global__ void __launch_bounds__(256) ln_kernel(
    const float* __restrict__ T, const float* __restrict__ X /*resid, or null*/,
    const float* __restrict__ g, const float* __restrict__ bb,
    float* __restrict__ outF, _Float16* __restrict__ outH /*or null*/) {
  int row = blockIdx.x * 8 + (threadIdx.x >> 5);
  int lane = threadIdx.x & 31;
  float v[8];
#pragma unroll
  for (int i = 0; i < 8; ++i) {
    int idx = i * 32 + lane;
    float x = T[(size_t)row * DM + idx];
    if (X) x += X[(size_t)row * DM + idx];
    v[i] = x;
  }
  float s = 0.f;
#pragma unroll
  for (int i = 0; i < 8; ++i) s += v[i];
#pragma unroll
  for (int m = 1; m < 32; m <<= 1) s += __shfl_xor(s, m, 32);
  float mean = s * (1.f / DM);
  float q = 0.f;
#pragma unroll
  for (int i = 0; i < 8; ++i) { float d = v[i] - mean; q += d * d; }
#pragma unroll
  for (int m = 1; m < 32; m <<= 1) q += __shfl_xor(q, m, 32);
  float rstd = rsqrtf(q * (1.f / DM) + 1e-5f);
#pragma unroll
  for (int i = 0; i < 8; ++i) {
    int idx = i * 32 + lane;
    float y = (v[i] - mean) * rstd * g[idx] + bb[idx];
    outF[(size_t)row * DM + idx] = y;
    if (outH) outH[(size_t)row * DM + idx] = (_Float16)y;
  }
}

// ------------------------------------------------------------------- driver
extern "C" void kernel_launch(void* const* d_in, const int* in_sizes, int n_in,
                              void* d_out, int out_size, void* d_ws,
                              size_t ws_size, hipStream_t stream) {
  (void)n_in; (void)out_size; (void)ws_size;
  const float* h_node = (const float*)d_in[0];
  const int*   batch  = (const int*)d_in[1];
  const float* qkv_w = (const float*)d_in[2];
  const float* qkv_b = (const float*)d_in[3];
  const float* out_w = (const float*)d_in[4];
  const float* out_b = (const float*)d_in[5];
  const float* ln1_g = (const float*)d_in[6];
  const float* ln1_b = (const float*)d_in[7];
  const float* ff1_w = (const float*)d_in[8];
  const float* ff1_b = (const float*)d_in[9];
  const float* ff2_w = (const float*)d_in[10];
  const float* ff2_b = (const float*)d_in[11];
  const float* ln2_g = (const float*)d_in[12];
  const float* ln2_b = (const float*)d_in[13];
  const float* lnf_g = (const float*)d_in[14];
  const float* lnf_b = (const float*)d_in[15];
  const int total = in_sizes[0] / DM;

  char* ws = (char*)d_ws;
  size_t off = 0;
  auto carve = [&](size_t bytes) {
    void* p = ws + off;
    off = (off + bytes + 255) & ~(size_t)255;
    return p;
  };
  int*       counts = (int*)carve(512);
  int*       starts = (int*)carve(512);
  float*     X    = (float*)carve((size_t)NTOK * DM * 4);       // activations
  _Float16*  Xh   = (_Float16*)carve((size_t)NTOK * DM * 2);    // f16 copy
  float*     T    = (float*)carve((size_t)NTOK * DM * 4);       // pre-LN temp
  _Float16*  QKVU = (_Float16*)carve((size_t)NTOK * 1024 * 2);  // qkv / ffn-hid
  _Float16*  Oh   = (_Float16*)carve((size_t)NTOK * DM * 2);    // attn out
  _Float16*  Wqkv = (_Float16*)carve((size_t)2 * 3 * DM * DM * 2);
  _Float16*  Wout = (_Float16*)carve((size_t)2 * DM * DM * 2);
  _Float16*  Wff1 = (_Float16*)carve((size_t)2 * FF * DM * 2);
  _Float16*  Wff2 = (_Float16*)carve((size_t)2 * DM * FF * 2);

  // pad/scatter
  zero_i32<<<1, 128, 0, stream>>>(counts, NB);
  zero_f32<<<2048, 256, 0, stream>>>(X, (size_t)NTOK * DM);
  zero_f32<<<1024, 256, 0, stream>>>((float*)Xh, (size_t)NTOK * DM / 2);
  count_kernel<<<(total + 255) / 256, 256, 0, stream>>>(batch, counts, total);
  scan_kernel<<<1, 32, 0, stream>>>(counts, starts);
  scatter_kernel<<<(total * DM + 255) / 256, 256, 0, stream>>>(
      h_node, batch, counts, starts, X, Xh, total);
  // f16 weights
  cvt_f16<<<512, 256, 0, stream>>>(qkv_w, Wqkv, (size_t)2 * 3 * DM * DM);
  cvt_f16<<<512, 256, 0, stream>>>(out_w, Wout, (size_t)2 * DM * DM);
  cvt_f16<<<512, 256, 0, stream>>>(ff1_w, Wff1, (size_t)2 * FF * DM);
  cvt_f16<<<512, 256, 0, stream>>>(ff2_w, Wff2, (size_t)2 * DM * FF);

  for (int l = 0; l < 2; ++l) {
    // QKV projection: [32768,256] x [768,256]^T -> f16 [32768,768]
    gemm_wmma<true, false><<<dim3(256, 6), 256, 0, stream>>>(
        Xh, Wqkv + (size_t)l * 3 * DM * DM, qkv_b + l * 3 * DM, nullptr, QKVU,
        NTOK, 3 * DM, DM);
    attn_kernel<<<NB * HEADS, 128, 0, stream>>>(QKVU, counts, Oh);
    // output projection -> f32 T
    gemm_wmma<false, false><<<dim3(256, 2), 256, 0, stream>>>(
        Oh, Wout + (size_t)l * DM * DM, out_b + l * DM, T, nullptr, NTOK, DM,
        DM);
    ln_kernel<<<4096, 256, 0, stream>>>(T, X, ln1_g + l * DM, ln1_b + l * DM, X,
                                        Xh);
    // FFN
    gemm_wmma<true, true><<<dim3(256, 8), 256, 0, stream>>>(
        Xh, Wff1 + (size_t)l * FF * DM, ff1_b + l * FF, nullptr, QKVU, NTOK, FF,
        DM);
    gemm_wmma<false, false><<<dim3(256, 2), 256, 0, stream>>>(
        QKVU, Wff2 + (size_t)l * DM * FF, ff2_b + l * DM, T, nullptr, NTOK, DM,
        FF);
    ln_kernel<<<4096, 256, 0, stream>>>(T, X, ln2_g + l * DM, ln2_b + l * DM, X,
                                        Xh);
  }
  ln_kernel<<<4096, 256, 0, stream>>>(X, nullptr, lnf_g, lnf_b, (float*)d_out,
                                      nullptr);
}
